// CausalScaledDotProductAttention_5944234737841
// MI455X (gfx1250) — matrix operand order repaired and checked by reading, not compiled, not run on hardware
//
#include <hip/hip_runtime.h>
#include <hip/hip_bf16.h>
#include <cstdint>

typedef __attribute__((ext_vector_type(16))) __bf16 v16bf;
typedef __attribute__((ext_vector_type(8)))  float  v8f;

union BF16x16 { v16bf v; unsigned short u[16]; unsigned int d[8]; };

// native bf16 convert (single hardware cvt, not the 4-op emulation)
__device__ __forceinline__ unsigned short f2bf(float x) {
    union { __bf16 h; unsigned short u; } c;
    c.h = (__bf16)x;
    return c.u;
}

__device__ __forceinline__ v8f wmma_bf16(const BF16x16& a, const BF16x16& b, v8f c) {
    return __builtin_amdgcn_wmma_f32_16x16x32_bf16(
        false, a.v, false, b.v, (short)0, c, false, false);
}

// A-fragment K offset (contiguous K order, matches B): VGPR i, half = lane>>4
__device__ __forceinline__ int a_koff(int i, int half) {
    return 2 * i + 16 * half;
}

// ---------------------------------------------------------------------------
// Pre-conversion kernels (bandwidth-bound, run once)
// ---------------------------------------------------------------------------
__global__ __launch_bounds__(256) void cvt_f32_to_bf16(
    const float4* __restrict__ src, uint2* __restrict__ dst, int n4)
{
    int i = blockIdx.x * 256 + threadIdx.x;
    if (i < n4) {
        float4 v = src[i];
        uint2 o;
        o.x = (unsigned)f2bf(v.x) | ((unsigned)f2bf(v.y) << 16);
        o.y = (unsigned)f2bf(v.z) | ((unsigned)f2bf(v.w) << 16);
        dst[i] = o;
    }
}

// W [K,N] f32  ->  Wt [N,K] bf16   (so B-fragment K-pairs are contiguous)
__global__ __launch_bounds__(256) void transpose_cvt_bf16(
    const float* __restrict__ W, unsigned short* __restrict__ Wt, int K, int N)
{
    __shared__ float tile[32][33];
    const int n0 = blockIdx.x * 32;
    const int k0 = blockIdx.y * 32;
    const int tx = threadIdx.x;        // 0..31
    const int ty = threadIdx.y;        // 0..7
#pragma unroll
    for (int i = 0; i < 32; i += 8)
        tile[ty + i][tx] = W[(size_t)(k0 + ty + i) * N + (n0 + tx)];
    __syncthreads();
#pragma unroll
    for (int i = 0; i < 32; i += 8)
        Wt[(size_t)(n0 + ty + i) * K + (k0 + tx)] = f2bf(tile[tx][ty + i]);
}

// ---------------------------------------------------------------------------
// GEMM: out[bf16] = A[bf16, MxK] * Bt[bf16, NxK]^T + bias
// Each wave computes a 16x64 strip (A fragment reused across 4 WMMAs).
// TRANS stores as [B, N, S=2048] (for V^T), else [M, N].
// ---------------------------------------------------------------------------
template <bool TRANS>
__global__ __launch_bounds__(256) void gemm_bf16(
    const unsigned short* __restrict__ A,   // [M, K]
    const unsigned short* __restrict__ Bt,  // [N, K]
    const float* __restrict__ bias,
    unsigned short* __restrict__ out,
    int M, int K, int N)
{
    const int tid  = threadIdx.x;
    const int lane = tid & 31;
    const int wave = tid >> 5;
    const int half = lane >> 4;
    const int l16  = lane & 15;

    const int ntN = N >> 6;                       // 64-wide N tiles
    const long wid = (long)blockIdx.x * 8 + wave;
    const int tN = (int)(wid % ntN);
    const int tM = (int)(wid / ntN);

    const int am = tM * 16 + l16;
    const unsigned short* Arow = A + (size_t)am * K;
    const unsigned short* Brow[4];
#pragma unroll
    for (int t = 0; t < 4; ++t)
        Brow[t] = Bt + (size_t)(tN * 64 + t * 16 + l16) * K;

    v8f acc[4];
    {
        v8f z = {};
#pragma unroll
        for (int t = 0; t < 4; ++t) acc[t] = z;
    }

    for (int kk = 0; kk < K; kk += 32) {
        BF16x16 a;
#pragma unroll
        for (int i = 0; i < 8; ++i)
            a.d[i] = *(const unsigned int*)&Arow[a_koff(i, half) + kk];
#pragma unroll
        for (int t = 0; t < 4; ++t) {
            BF16x16 b;
#pragma unroll
            for (int j = 0; j < 8; ++j)
                b.d[j] = *(const unsigned int*)&Brow[t][kk + 2 * j + 16 * half];
            acc[t] = wmma_bf16(a, b, acc[t]);
        }
    }

#pragma unroll
    for (int t = 0; t < 4; ++t) {
        const int n = tN * 64 + t * 16 + l16;
        const float bcol = bias[n];
#pragma unroll
        for (int r = 0; r < 8; ++r) {
            const int m = tM * 16 + r + 8 * half;
            unsigned short obf = f2bf(acc[t][r] + bcol);
            if (TRANS) {   // [B, N, 2048]
                out[((size_t)((m >> 11) * N + n) << 11) + (m & 2047)] = obf;
            } else {
                out[(size_t)m * N + n] = obf;
            }
        }
    }
}

// ---------------------------------------------------------------------------
// Flash attention. One block = (batch, 16-query block); 8 waves.
// Each wave: one 16x16 score tile of the 16x128 key panel, then a 64-wide
// O-slice of P.V. Softmax stats parallelized across all 256 threads.
// ---------------------------------------------------------------------------
#define ATT_S 2048
#define ATT_D 1024
#define ATT_O 512

__global__ __launch_bounds__(256) void flash_attn(
    const unsigned short* __restrict__ Qb,   // [B*S, D] bf16
    const unsigned short* __restrict__ Kb,   // [B*S, D] bf16
    const unsigned short* __restrict__ Vt,   // [B, O, S] bf16
    float* __restrict__ Out)                 // [B, S, O] f32
{
    __shared__ unsigned short Qs[16 * ATT_D];     // 32 KB staged Q block
    __shared__ float          Sc[16][128];        // scaled/masked scores
    __shared__ unsigned short Pb[16][128];        // probs, bf16
    __shared__ float          Red[16][16];        // reduction scratch
    __shared__ float Mrow[16], Lrow[16], Frow[16];

    const float scale = 0.03125f;                 // 1/sqrt(1024)
    const int tid  = threadIdx.x;
    const int lane = tid & 31;
    const int wave = tid >> 5;
    const int half = lane >> 4;
    const int l16  = lane & 15;
    const int srow = tid >> 4;                    // softmax: row 0..15
    const int sseg = tid & 15;                    // softmax: 8-col segment

    const int qb = blockIdx.x % (ATT_S / 16);
    const int b  = blockIdx.x / (ATT_S / 16);
    const int q0 = qb * 16;

    {   // stage 16x1024 bf16 Q block (contiguous 8192 dwords)
        const unsigned int* src = (const unsigned int*)(Qb + ((size_t)b * ATT_S + q0) * ATT_D);
        unsigned int* dst = (unsigned int*)Qs;
        for (int i = tid; i < 16 * ATT_D / 2; i += 256) dst[i] = src[i];
    }
    if (tid < 16) { Mrow[tid] = -__builtin_inff(); Lrow[tid] = 0.0f; Frow[tid] = 1.0f; }
    __syncthreads();

    v8f accO[4];
    {
        v8f z = {};
#pragma unroll
        for (int t = 0; t < 4; ++t) accO[t] = z;
    }

    const int nkb = (q0 + 16 + 127) / 128;        // causal loop bound
    for (int it = 0; it < nkb; ++it) {
        const int j0 = it * 128;
        const int k0 = j0 + wave * 16;

        // ---- scores: 16xD . Dx16 ----
        v8f sacc = {};
        const unsigned short* Krow = Kb + ((size_t)b * ATT_S + (k0 + l16)) * ATT_D;
        for (int kk = 0; kk < ATT_D; kk += 32) {
            BF16x16 a, bm;
#pragma unroll
            for (int i = 0; i < 8; ++i)
                a.d[i] = *(const unsigned int*)&Qs[l16 * ATT_D + a_koff(i, half) + kk];
#pragma unroll
            for (int j = 0; j < 8; ++j)
                bm.d[j] = *(const unsigned int*)&Krow[kk + 2 * j + 16 * half];
            sacc = wmma_bf16(a, bm, sacc);
        }
#pragma unroll
        for (int r = 0; r < 8; ++r) {
            const int m  = r + 8 * half;
            const int kg = k0 + l16;
            float s = sacc[r] * scale;
            if (kg > q0 + m) s = -__builtin_inff();
            Sc[m][wave * 16 + l16] = s;
        }
        __syncthreads();

        // ---- parallel online softmax: partial max ----
        {
            float pm = -__builtin_inff();
#pragma unroll
            for (int c = 0; c < 8; ++c) pm = fmaxf(pm, Sc[srow][sseg * 8 + c]);
            Red[srow][sseg] = pm;
        }
        __syncthreads();
        if (sseg == 0) {
            float mx = Mrow[srow];
#pragma unroll
            for (int s2 = 0; s2 < 16; ++s2) mx = fmaxf(mx, Red[srow][s2]);
            float f = __expf(Mrow[srow] - mx);
            Mrow[srow] = mx; Frow[srow] = f; Lrow[srow] *= f;
        }
        __syncthreads();
        // ---- exp + P tile + partial sum ----
        {
            const float mx = Mrow[srow];
            float psum = 0.0f;
#pragma unroll
            for (int c = 0; c < 8; ++c) {
                float p = __expf(Sc[srow][sseg * 8 + c] - mx);
                Pb[srow][sseg * 8 + c] = f2bf(p);
                psum += p;
            }
            Red[srow][sseg] = psum;
        }
        __syncthreads();
        if (sseg == 0) {
            float l = Lrow[srow];
#pragma unroll
            for (int s2 = 0; s2 < 16; ++s2) l += Red[srow][s2];
            Lrow[srow] = l;
        }
        __syncthreads();

        // ---- rescale accumulators, O += P(16x128) . V(128x64-slice) ----
        float fr[8];
#pragma unroll
        for (int r = 0; r < 8; ++r) fr[r] = Frow[r + 8 * half];
#pragma unroll
        for (int t = 0; t < 4; ++t)
#pragma unroll
            for (int r = 0; r < 8; ++r) accO[t][r] *= fr[r];

        const int ob = wave * 64;
#pragma unroll
        for (int ks = 0; ks < 4; ++ks) {
            BF16x16 a;
#pragma unroll
            for (int i = 0; i < 8; ++i)
                a.d[i] = *(const unsigned int*)&Pb[l16][a_koff(i, half) + ks * 32];
#pragma unroll
            for (int t = 0; t < 4; ++t) {
                const int o = ob + t * 16 + l16;
                const unsigned short* Vrow = Vt + ((size_t)b * ATT_O + o) * ATT_S;
                BF16x16 bm;
#pragma unroll
                for (int j = 0; j < 8; ++j)
                    bm.d[j] = *(const unsigned int*)&Vrow[j0 + ks * 32 + 2 * j + 16 * half];
                accO[t] = wmma_bf16(a, bm, accO[t]);
            }
        }
        __syncthreads();
    }

    // ---- epilogue ----
    float linv[8];
#pragma unroll
    for (int r = 0; r < 8; ++r) linv[r] = 1.0f / Lrow[r + 8 * half];
#pragma unroll
    for (int t = 0; t < 4; ++t) {
#pragma unroll
        for (int r = 0; r < 8; ++r) {
            const int m = r + 8 * half;
            const int o = wave * 64 + t * 16 + l16;
            Out[((size_t)b * ATT_S + q0 + m) * ATT_O + o] = accO[t][r] * linv[r];
        }
    }
}

// ---------------------------------------------------------------------------
extern "C" void kernel_launch(void* const* d_in, const int* in_sizes, int n_in,
                              void* d_out, int out_size, void* d_ws, size_t ws_size,
                              hipStream_t stream) {
    const float* x  = (const float*)d_in[0];
    const float* Wq = (const float*)d_in[1];
    const float* bq = (const float*)d_in[2];
    const float* Wk = (const float*)d_in[3];
    const float* bk = (const float*)d_in[4];
    const float* Wv = (const float*)d_in[5];
    const float* bv = (const float*)d_in[6];
    float* out = (float*)d_out;

    const int B = 8, S = 2048, D = 1024, O = 512;
    const int M = B * S;   // 16384

    unsigned short* Xb  = (unsigned short*)d_ws;          // [M, D]      32 MB
    unsigned short* Wqt = Xb  + (size_t)M * D;            // [D, D]       2 MB
    unsigned short* Wkt = Wqt + (size_t)D * D;            // [D, D]       2 MB
    unsigned short* Wvt = Wkt + (size_t)D * D;            // [O, D]       1 MB
    unsigned short* Qb  = Wvt + (size_t)O * D;            // [M, D]      32 MB
    unsigned short* Kb  = Qb  + (size_t)M * D;            // [M, D]      32 MB
    unsigned short* Vt  = Kb  + (size_t)M * D;            // [B, O, S]   16 MB

    // pre-convert / transpose (bandwidth-bound)
    {
        const int n4 = M * D / 4;
        cvt_f32_to_bf16<<<(n4 + 255) / 256, 256, 0, stream>>>((const float4*)x, (uint2*)Xb, n4);
        dim3 tb(32, 8);
        transpose_cvt_bf16<<<dim3(D / 32, D / 32), tb, 0, stream>>>(Wq, Wqt, D, D);
        transpose_cvt_bf16<<<dim3(D / 32, D / 32), tb, 0, stream>>>(Wk, Wkt, D, D);
        transpose_cvt_bf16<<<dim3(O / 32, D / 32), tb, 0, stream>>>(Wv, Wvt, D, O);
    }

    // projections (WMMA)
    const int wQK = (M / 16) * (D / 64);   // 16384 waves -> 2048 blocks
    const int wV  = (M / 16) * (O / 64);   //  8192 waves -> 1024 blocks
    gemm_bf16<false><<<wQK / 8, 256, 0, stream>>>(Xb, Wqt, bq, Qb, M, D, D);
    gemm_bf16<false><<<wQK / 8, 256, 0, stream>>>(Xb, Wkt, bk, Kb, M, D, D);
    gemm_bf16<true ><<<wV  / 8, 256, 0, stream>>>(Xb, Wvt, bv, Vt, M, D, O);

    // attention (WMMA flash)
    flash_attn<<<B * (S / 16), 256, 0, stream>>>(Qb, Kb, Vt, out);
}
